// ClassicalSelfAttention_65481071398957
// MI455X (gfx1250) — compile-verified, matrix operand order checked by reading, not run on hardware
//
#include <hip/hip_runtime.h>

// Problem constants
#define B_ 4
#define N_ 2048
#define E_ 1024
#define H_ 16
#define D_ 64

typedef __attribute__((ext_vector_type(16))) __bf16 v16bf;
typedef __attribute__((ext_vector_type(8)))  float  v8f;
typedef __attribute__((ext_vector_type(4)))  unsigned v4u;
typedef __attribute__((ext_vector_type(8)))  int      v8i_;
typedef __attribute__((ext_vector_type(4)))  int      v4i_;

#ifndef __has_builtin
#define __has_builtin(x) 0
#endif
#if __has_builtin(__builtin_amdgcn_tensor_load_to_lds)
#define HAVE_TDM 1
#endif
#if __has_include(<hip/amd_detail/amd_gfx1250_TDM.h>)
#define TDM_6ARG 1
#endif

__device__ __forceinline__ unsigned short f2bf(float f) {
  unsigned u = __builtin_bit_cast(unsigned, f);
  u += 0x7FFFu + ((u >> 16) & 1u);        // round-to-nearest-even
  return (unsigned short)(u >> 16);
}

union FragU { v16bf v; uint2 d[4]; };

// 16x32 bf16 fragment (A-layout / B^T-layout) from row-major storage.
// Lane L (0-15): row L, K = {0..7,16..23}; lanes 16-31: K = {8..15,24..31}.
__device__ __forceinline__ v16bf load_frag(const unsigned short* base, int stride) {
  const int lane = threadIdx.x & 31;
  const unsigned short* p = base + (lane & 15) * stride + ((lane >> 4) << 3);
  FragU f;
  f.d[0] = *(const uint2*)(p);
  f.d[1] = *(const uint2*)(p + 4);
  f.d[2] = *(const uint2*)(p + 16);
  f.d[3] = *(const uint2*)(p + 20);
  return f.v;
}

__device__ __forceinline__ v8f wmma_bf16(v16bf a, v16bf b, v8f c) {
  return __builtin_amdgcn_wmma_f32_16x16x32_bf16(false, a, false, b, (short)0, c,
                                                 false, false);
}

#if defined(HAVE_TDM)
// TDM 2D tile load: rows x 32 bf16 elements, row stride `stride_e` elements,
// LDS rows padded to 40 elements (pad_interval=16 DW, pad_amount=4 DW).
__device__ __forceinline__ void tdm_load_2d(const void* gptr, unsigned lds_off,
                                            unsigned tile_d0, unsigned tile_d1,
                                            unsigned stride_e, unsigned tdim0,
                                            unsigned tdim1) {
  unsigned long long ga = (unsigned long long)(uintptr_t)gptr;
  v4u g0;
  g0.x = 1u;                                   // count=1, user mode
  g0.y = lds_off;                              // LDS byte address
  g0.z = (unsigned)ga;                         // global addr [31:0]
  g0.w = (unsigned)(ga >> 32) | 0x80000000u;   // addr[56:32] | type=2
  v8i_ g1;
  g1[0] = (int)((1u << 16)                     // data_size = 2 bytes
                | (1u << 20)                   // pad_enable
                | (3u << 22)                   // pad_interval: 16 DWORDs
                | (3u << 25));                 // pad_amount: 4 DWORDs
  g1[1] = (int)((tdim0 & 0xFFFFu) << 16);
  g1[2] = (int)(((tdim0 >> 16) & 0xFFFFu) | ((tdim1 & 0xFFFFu) << 16));
  g1[3] = (int)(((tdim1 >> 16) & 0xFFFFu) | (tile_d0 << 16));
  g1[4] = (int)(tile_d1 & 0xFFFFu);            // tile_dim1 (tile_dim2 = 0)
  g1[5] = (int)stride_e;                       // tensor_dim0_stride
  g1[6] = 0;
  g1[7] = 0;
  v4i_ z4 = {0, 0, 0, 0};
#if defined(TDM_6ARG)
  v8i_ z8 = {0, 0, 0, 0, 0, 0, 0, 0};
  __builtin_amdgcn_tensor_load_to_lds(g0, g1, z4, z4, z8, 0);
#else
  __builtin_amdgcn_tensor_load_to_lds(g0, g1, z4, z4, 0);
#endif
}
#endif

// ---------------------------------------------------------------------------
// Kernel 1: cast f32 -> bf16 with transpose. in: R x C, out: C x R.
// ---------------------------------------------------------------------------
__global__ void k_cast_transpose(const float* __restrict__ in,
                                 unsigned short* __restrict__ out,
                                 int R, int C) {
  long long tid = (long long)blockIdx.x * blockDim.x + threadIdx.x;
  long long tot = (long long)R * C;
  if (tid >= tot) return;
  int r = (int)(tid / C), c = (int)(tid % C);
  out[(long long)c * R + r] = f2bf(in[tid]);
}

// ---------------------------------------------------------------------------
// Kernel 2: QKV GEMM.  x(8192x1024 f32) @ Wqkv -> scatter Q,K (N-major) and
// V transposed (D-major) as bf16. Wt is Wqkv^T (3072x1024 bf16).
// Workgroup tile 128(M) x 128(N), 8 waves in 4x2, each wave 32x64.
// B tile staged by the Tensor Data Mover; A tile converted f32->bf16 by VALU.
// ---------------------------------------------------------------------------
__global__ void __launch_bounds__(256)
k_qkv_gemm(const float* __restrict__ x, const unsigned short* __restrict__ Wt,
           unsigned short* __restrict__ Q, unsigned short* __restrict__ K,
           unsigned short* __restrict__ Vt) {
  __shared__ __attribute__((aligned(16))) unsigned short As[128 * 40];
  __shared__ __attribute__((aligned(16))) unsigned short Bs[128 * 40];
  const int t = threadIdx.x;
  const int wave = t >> 5, lane = t & 31;
  const int wm = wave & 3, wn = wave >> 2;
  const int g0 = blockIdx.y * 128;                // output row base
  const int n0 = blockIdx.x * 128;                // output col base

  v8f acc[2][4] = {};

  for (int k0 = 0; k0 < E_; k0 += 32) {
    // Stage B: 128x32 bf16 rows of Wt (async tensor DMA when available)
#if defined(HAVE_TDM)
    if (wave == 0)
      tdm_load_2d(Wt + (long long)n0 * E_ + k0,
                  (unsigned)(uintptr_t)&Bs[0], 32, 128, E_, E_, 3 * E_);
#else
    for (int i = 0; i < 4; i++) {
      int lin = t + i * 256;
      int row = lin >> 3, c4 = (lin & 7) << 2;
      *(uint2*)(Bs + row * 40 + c4) =
          *(const uint2*)(Wt + (long long)(n0 + row) * E_ + k0 + c4);
    }
#endif
    // Stage A: 128x32 f32 -> bf16 LDS (overlaps with the TDM copy)
    for (int i = 0; i < 4; i++) {
      int lin = t + i * 256;                       // 0..1023
      int row = lin >> 3, c4 = (lin & 7) << 2;
      const float4 f =
          *(const float4*)(x + (long long)(g0 + row) * E_ + k0 + c4);
      uint2 p;
      p.x = (unsigned)f2bf(f.x) | ((unsigned)f2bf(f.y) << 16);
      p.y = (unsigned)f2bf(f.z) | ((unsigned)f2bf(f.w) << 16);
      *(uint2*)(As + row * 40 + c4) = p;
    }
#if defined(HAVE_TDM)
    if (wave == 0) __builtin_amdgcn_s_wait_tensorcnt(0);
#endif
    __syncthreads();
    v16bf a0 = load_frag(As + (wm * 32 + 0) * 40, 40);
    v16bf a1 = load_frag(As + (wm * 32 + 16) * 40, 40);
    v16bf bf[4];
    for (int u = 0; u < 4; u++)
      bf[u] = load_frag(Bs + (wn * 64 + u * 16) * 40, 40);
    for (int u = 0; u < 4; u++) {
      acc[0][u] = wmma_bf16(a0, bf[u], acc[0][u]);
      acc[1][u] = wmma_bf16(a1, bf[u], acc[1][u]);
    }
    __syncthreads();
  }

  // Epilogue: scatter into Q (scaled by 1/sqrt(D)=0.125), K, V^T
  for (int ti = 0; ti < 2; ti++)
    for (int tj = 0; tj < 4; tj++) {
      int cb  = n0 + wn * 64 + tj * 16;            // 16-aligned column base
      int mat = cb >> 10;                          // 0=Q 1=K 2=V
      int h   = (cb & 1023) >> 6;
      int d   = (cb & 63) + (lane & 15);
      for (int j = 0; j < 8; j++) {
        int g = g0 + wm * 32 + ti * 16 + j + ((lane >> 4) << 3);
        int b = g >> 11, n = g & (N_ - 1);
        float v = acc[ti][tj][j];
        long long bh = (long long)(b * H_ + h);
        if (mat == 0)
          Q[(bh * N_ + n) * D_ + d] = f2bf(v * 0.125f);
        else if (mat == 1)
          K[(bh * N_ + n) * D_ + d] = f2bf(v);
        else
          Vt[(bh * D_ + d) * N_ + n] = f2bf(v);
      }
    }
}

// ---------------------------------------------------------------------------
// Kernel 3: flash attention. grid=(N/128, B*H), 8 waves; wave owns 16 q-rows.
// ---------------------------------------------------------------------------
__global__ void __launch_bounds__(256)
k_attn(const unsigned short* __restrict__ Q, const unsigned short* __restrict__ K,
       const unsigned short* __restrict__ Vt, unsigned short* __restrict__ O) {
  __shared__ __attribute__((aligned(16))) unsigned short Ps[8][16 * 40];
  const int t = threadIdx.x, wave = t >> 5, lane = t & 31;
  const int bh = blockIdx.y;
  const int q0 = blockIdx.x * 128 + wave * 16;
  const unsigned short* Qp = Q + (long long)bh * N_ * D_;
  const unsigned short* Kp = K + (long long)bh * N_ * D_;
  const unsigned short* Vp = Vt + (long long)bh * D_ * N_;

  v16bf qf0 = load_frag(Qp + q0 * D_ + 0, D_);
  v16bf qf1 = load_frag(Qp + q0 * D_ + 32, D_);

  v8f o[4] = {};
  float mrow[8], lrow[8];
  for (int j = 0; j < 8; j++) { mrow[j] = -1e30f; lrow[j] = 0.f; }
  unsigned short* Pw = &Ps[wave][0];

  for (int kv = 0; kv < N_; kv += 32) {
    v8f s0 = {}, s1 = {};
    {
      v16bf k00 = load_frag(Kp + (kv + 0) * D_ + 0, D_);
      v16bf k10 = load_frag(Kp + (kv + 16) * D_ + 0, D_);
      s0 = wmma_bf16(qf0, k00, s0);
      s1 = wmma_bf16(qf0, k10, s1);
      v16bf k01 = load_frag(Kp + (kv + 0) * D_ + 32, D_);
      v16bf k11 = load_frag(Kp + (kv + 16) * D_ + 32, D_);
      s0 = wmma_bf16(qf1, k01, s0);
      s1 = wmma_bf16(qf1, k11, s1);
    }
    // Online softmax (C-layout: VGPR j <-> row j + 8*(lane>=16))
    for (int j = 0; j < 8; j++) {
      float tm = fmaxf(s0[j], s1[j]);
      for (int off = 8; off; off >>= 1) tm = fmaxf(tm, __shfl_xor(tm, off, 32));
      float nm   = fmaxf(mrow[j], tm);
      float corr = __expf(mrow[j] - nm);
      float p0   = __expf(s0[j] - nm);
      float p1   = __expf(s1[j] - nm);
      float rs   = p0 + p1;
      for (int off = 8; off; off >>= 1) rs += __shfl_xor(rs, off, 32);
      lrow[j] = lrow[j] * corr + rs;
      mrow[j] = nm;
      for (int u = 0; u < 4; u++) o[u][j] *= corr;
      int row = j + ((lane >> 4) << 3);
      int col = lane & 15;
      Pw[row * 40 + col]      = f2bf(p0);
      Pw[row * 40 + col + 16] = f2bf(p1);
    }
    v16bf pa = load_frag(Pw, 40);   // same-wave LDS: in-order
    for (int u = 0; u < 4; u++) {
      v16bf bv = load_frag(Vp + (u * 16) * N_ + kv, N_);
      o[u] = wmma_bf16(pa, bv, o[u]);
    }
  }

  const int b = bh >> 4, h = bh & 15;
  for (int u = 0; u < 4; u++)
    for (int j = 0; j < 8; j++) {
      int qrow = q0 + j + ((lane >> 4) << 3);
      float v = o[u][j] / lrow[j];
      O[((long long)(b * N_ + qrow)) * E_ + h * D_ + u * 16 + (lane & 15)] =
          f2bf(v);
    }
}

// ---------------------------------------------------------------------------
// Kernel 4: output projection GEMM + bias. Both A and B tiles staged by TDM.
// Workgroup tile 128x128, per-wave 32x64.
// ---------------------------------------------------------------------------
__global__ void __launch_bounds__(256)
k_out_gemm(const unsigned short* __restrict__ A,
           const unsigned short* __restrict__ Wt,
           const float* __restrict__ bias, float* __restrict__ out) {
  __shared__ __attribute__((aligned(16))) unsigned short As[128 * 40];
  __shared__ __attribute__((aligned(16))) unsigned short Bs[128 * 40];
  const int t = threadIdx.x, wave = t >> 5, lane = t & 31;
  const int wm = wave & 3, wn = wave >> 2;
  const int g0 = blockIdx.y * 128, n0 = blockIdx.x * 128;
  v8f acc[2][4] = {};

  for (int k0 = 0; k0 < E_; k0 += 32) {
#if defined(HAVE_TDM)
    if (wave == 0) {
      tdm_load_2d(A + (long long)g0 * E_ + k0,
                  (unsigned)(uintptr_t)&As[0], 32, 128, E_, E_, B_ * N_);
      tdm_load_2d(Wt + (long long)n0 * E_ + k0,
                  (unsigned)(uintptr_t)&Bs[0], 32, 128, E_, E_, E_);
      __builtin_amdgcn_s_wait_tensorcnt(0);
    }
#else
    for (int i = 0; i < 4; i++) {
      int lin = t + i * 256;
      int row = lin >> 3, c4 = (lin & 7) << 2;
      *(uint2*)(As + row * 40 + c4) =
          *(const uint2*)(A + (long long)(g0 + row) * E_ + k0 + c4);
      *(uint2*)(Bs + row * 40 + c4) =
          *(const uint2*)(Wt + (long long)(n0 + row) * E_ + k0 + c4);
    }
#endif
    __syncthreads();
    v16bf a0 = load_frag(As + (wm * 32 + 0) * 40, 40);
    v16bf a1 = load_frag(As + (wm * 32 + 16) * 40, 40);
    v16bf bf[4];
    for (int u = 0; u < 4; u++)
      bf[u] = load_frag(Bs + (wn * 64 + u * 16) * 40, 40);
    for (int u = 0; u < 4; u++) {
      acc[0][u] = wmma_bf16(a0, bf[u], acc[0][u]);
      acc[1][u] = wmma_bf16(a1, bf[u], acc[1][u]);
    }
    __syncthreads();
  }

  for (int ti = 0; ti < 2; ti++)
    for (int tj = 0; tj < 4; tj++) {
      int c = n0 + wn * 64 + tj * 16 + (lane & 15);
      float bv = bias[c];
      for (int j = 0; j < 8; j++) {
        int g = g0 + wm * 32 + ti * 16 + j + ((lane >> 4) << 3);
        out[(long long)g * E_ + c] = acc[ti][tj][j] + bv;
      }
    }
}

// ---------------------------------------------------------------------------
extern "C" void kernel_launch(void* const* d_in, const int* in_sizes, int n_in,
                              void* d_out, int out_size, void* d_ws,
                              size_t ws_size, hipStream_t stream) {
  (void)in_sizes; (void)n_in; (void)out_size; (void)ws_size;
  const float* x    = (const float*)d_in[0];
  const float* Wqkv = (const float*)d_in[1];
  const float* Wout = (const float*)d_in[2];
  const float* bout = (const float*)d_in[3];
  float* out = (float*)d_out;

  // Workspace layout (bf16 as unsigned short), ~72 MB total
  unsigned short* ws = (unsigned short*)d_ws;
  size_t off = 0;
  unsigned short* Wqkv_t = ws + off; off += (size_t)(3 * E_) * E_;   // 3072x1024
  unsigned short* Wout_t = ws + off; off += (size_t)E_ * E_;         // 1024x1024
  unsigned short* Qb = ws + off; off += (size_t)B_ * H_ * N_ * D_;
  unsigned short* Kb = ws + off; off += (size_t)B_ * H_ * N_ * D_;
  unsigned short* Vt = ws + off; off += (size_t)B_ * H_ * D_ * N_;
  unsigned short* Ob = ws + off; off += (size_t)B_ * N_ * E_;

  {
    long long tot = (long long)E_ * (3 * E_);
    k_cast_transpose<<<dim3((unsigned)((tot + 255) / 256)), dim3(256), 0,
                       stream>>>(Wqkv, Wqkv_t, E_, 3 * E_);
  }
  {
    long long tot = (long long)E_ * E_;
    k_cast_transpose<<<dim3((unsigned)((tot + 255) / 256)), dim3(256), 0,
                       stream>>>(Wout, Wout_t, E_, E_);
  }
  k_qkv_gemm<<<dim3(3 * E_ / 128, B_ * N_ / 128), dim3(256), 0, stream>>>(
      x, Wqkv_t, Qb, Kb, Vt);
  k_attn<<<dim3(N_ / 128, B_ * H_), dim3(256), 0, stream>>>(Qb, Kb, Vt, Ob);
  k_out_gemm<<<dim3(E_ / 128, B_ * N_ / 128), dim3(256), 0, stream>>>(
      Ob, Wout_t, bout, out);
}
